// ModulatedConv2d_20289425506659
// MI455X (gfx1250) — matrix operand order repaired and checked by reading, not compiled, not run on hardware
//
#include <hip/hip_runtime.h>
#include <hip/hip_bf16.h>

typedef __attribute__((ext_vector_type(16))) _Float16 v16h;
typedef __attribute__((ext_vector_type(8)))  _Float16 v8h;
typedef _Float16 v8h_u __attribute__((ext_vector_type(8), aligned(2)));
typedef __attribute__((ext_vector_type(8)))  float    v8f;

#define B_    16
#define CIN_  512
#define COUT_ 512
#define LDIM_ 512
#define HW_   32
#define NPIX  1024          // 32*32
#define KTOT  4608          // 512*9
#define TKS   64            // K per LDS stage (2 WMMA chunks of 32)
#define PHW   34            // padded 34x34
#define PIMG  1156          // 34*34

// ---- CDNA5 async copy helpers (ISA 15.18.3 / 9.6) --------------------------
typedef __attribute__((address_space(3))) _Float16 lds_h;

static __device__ __forceinline__ unsigned lds_off(const _Float16* p) {
    return (unsigned)(size_t)(lds_h*)(void*)const_cast<_Float16*>(p);
}
static __device__ __forceinline__ void async_copy_b128(const _Float16* g, _Float16* l) {
    unsigned lo = lds_off(l);
    unsigned long long ga = (unsigned long long)(size_t)g;
    asm volatile("global_load_async_to_lds_b128 %0, %1, off"
                 :: "v"(lo), "v"(ga) : "memory");
}
static __device__ __forceinline__ void wait_async0() {
    asm volatile("s_wait_asynccnt 0x0" ::: "memory");
}
static __device__ __forceinline__ v16h mkfrag(v8h lo, v8h hi) {
    return __builtin_shufflevector(lo, hi,
        0, 1, 2, 3, 4, 5, 6, 7, 8, 9, 10, 11, 12, 13, 14, 15);
}

// ---------------------------------------------------------------------------
// Kernel 1: modulation MLP  s = LeakyReLU(LN(w @ mod_w.T + mod_b)) * scale
// ---------------------------------------------------------------------------
__global__ __launch_bounds__(512) void k_modulate(
    const float* __restrict__ w, const float* __restrict__ mod_w,
    const float* __restrict__ mod_b, const float* __restrict__ ln_g,
    const float* __restrict__ ln_b, const float* __restrict__ scale,
    float* __restrict__ style)
{
    const int b = blockIdx.x;
    const int c = threadIdx.x;
    __shared__ float sw[LDIM_];
    __shared__ float red[CIN_];

    sw[c] = w[b * LDIM_ + c];
    __syncthreads();

    const float* mrow = mod_w + (size_t)c * LDIM_;
    float dot = 0.f;
    #pragma unroll 4
    for (int l = 0; l < LDIM_; ++l) dot += sw[l] * mrow[l];
    float s = dot + mod_b[c];

    red[c] = s;
    __syncthreads();
    for (int off = CIN_ / 2; off > 0; off >>= 1) {
        if (c < off) red[c] += red[c + off];
        __syncthreads();
    }
    float mu = red[0] * (1.0f / CIN_);
    __syncthreads();

    red[c] = s * s;
    __syncthreads();
    for (int off = CIN_ / 2; off > 0; off >>= 1) {
        if (c < off) red[c] += red[c + off];
        __syncthreads();
    }
    float var = red[0] * (1.0f / CIN_) - mu * mu;

    float t = (s - mu) * rsqrtf(var + 1e-5f) * ln_g[c] + ln_b[c];
    t = (t >= 0.f) ? t : 0.2f * t;
    style[b * CIN_ + c] = t * scale[c];
}

// ---------------------------------------------------------------------------
// Kernel 2: x (f32) -> zero-padded f16 image [b][cin][34][34]
// ---------------------------------------------------------------------------
__global__ __launch_bounds__(256) void k_pad(
    const float* __restrict__ x, _Float16* __restrict__ xp, int total)
{
    int i = blockIdx.x * 256 + threadIdx.x;
    if (i >= total) return;
    int xx = i % PHW;
    int t  = i / PHW;
    int yy = t % PHW;
    int bc = t / PHW;
    _Float16 v = (_Float16)0.f;
    if (xx >= 1 && xx <= HW_ && yy >= 1 && yy <= HW_)
        v = (_Float16)x[(size_t)bc * NPIX + (yy - 1) * HW_ + (xx - 1)];
    xp[i] = v;
}

// ---------------------------------------------------------------------------
// Kernel 3: per (b,cout): demodulate and pack f16 -> wmod[b][cout][4608]
// ---------------------------------------------------------------------------
__global__ __launch_bounds__(256) void k_demod(
    const float* __restrict__ weight, const float* __restrict__ style,
    _Float16* __restrict__ wmodh)
{
    const int blk  = blockIdx.x;
    const int b    = blk >> 9;
    const int cout = blk & 511;
    const int t    = threadIdx.x;

    const float* wrow = weight + (size_t)cout * KTOT;
    const float* st   = style + b * CIN_;

    float vals[18];
    float ss = 0.f;
    #pragma unroll
    for (int j = 0; j < 18; ++j) {
        int i = t + j * 256;
        float v = wrow[i] * st[i / 9];
        vals[j] = v;
        ss += v * v;
    }

    __shared__ float red[256];
    red[t] = ss;
    __syncthreads();
    for (int off = 128; off > 0; off >>= 1) {
        if (t < off) red[t] += red[t + off];
        __syncthreads();
    }
    float d = rsqrtf(red[0] + 1e-8f);

    _Float16* orow = wmodh + (size_t)blk * KTOT;
    #pragma unroll
    for (int j = 0; j < 18; ++j)
        orow[t + j * 256] = (_Float16)(vals[j] * d);
}

// ---------------------------------------------------------------------------
// Kernel 4: implicit-GEMM conv, v_wmma_f32_16x16x32_f16
//   block tile 128 (cout) x 64 (pix); 8 wave32 each owning 32x32
//   K staged 64 per double-buffered LDS stage -> 8 WMMA per wave per barrier
//   A tile via global_load_async_to_lds_b128; B tile reg-staged im2col
// grid = (NPIX/64, COUT_/128, B)
// ---------------------------------------------------------------------------
__global__ __launch_bounds__(256) void k_conv_wmma(
    const _Float16* __restrict__ wmodh, const _Float16* __restrict__ xp,
    float* __restrict__ out)
{
    const int nt   = blockIdx.x;
    const int mt   = blockIdx.y;
    const int b    = blockIdx.z;
    const int tid  = threadIdx.x;
    const int lane = tid & 31;
    const int wv   = tid >> 5;
    const int mi   = wv & 3;              // 32-row sub-tile
    const int ni   = wv >> 2;             // 32-col sub-tile

    __shared__ __align__(32) _Float16 lA[2][128 * TKS];  // [m][k]  16KB each
    __shared__ __align__(32) _Float16 lB[2][64 * TKS];   // [n][k]   8KB each

    const int cout0 = mt * 128;
    const int p0    = nt * 64;
    const int y0    = p0 >> 5;            // 2 image rows per pixel tile

    const _Float16* wbase = wmodh + (size_t)(b * COUT_ + cout0) * KTOT;
    const _Float16* xpb   = xp + (size_t)b * CIN_ * PIMG;

    v8f acc00 = {}, acc01 = {}, acc10 = {}, acc11 = {};

    // A fill: thread -> row tid>>1, 32-half group (tid&1) -> 4x b128 async
    const int ar = tid >> 1;
    const int ac = (tid & 1) * 32;
    // B fill: thread -> k-rows {br, br+32}, 8-pixel group (tid&7)*8
    const int br = tid >> 3;
    const int pg = (tid & 7) * 8;
    const int prow = y0 + (pg >> 5);      // padded image row (before +ky)
    const int pcol = (pg & 31);

    // incremental (cin, kidx) state for the NEXT stage to prefetch;
    // K advances by 64 == 1 (mod 9) per stage
    int cinA = br / 9,        kidxA = br - (br / 9) * 9;
    int cinB = (br + 32) / 9, kidxB = (br + 32) - ((br + 32) / 9) * 9;

    auto fillA = [&](int buf, int k0) {
        const _Float16* src = wbase + (size_t)ar * KTOT + k0 + ac;
        _Float16* dst = &lA[buf][ar * TKS + ac];
        async_copy_b128(src,      dst);
        async_copy_b128(src + 8,  dst + 8);
        async_copy_b128(src + 16, dst + 16);
        async_copy_b128(src + 24, dst + 24);
    };
    auto loadRow = [&](int cin, int kidx) -> v8h_u {
        int ky = (kidx * 11) >> 5;        // kidx/3 for 0..8
        int kx = kidx - ky * 3;
        const _Float16* s = xpb + (size_t)cin * PIMG
                          + (size_t)(prow + ky) * PHW + pcol + kx;
        return *(const v8h_u*)s;
    };
    auto advance = [](int& cin, int& kidx) {
        if (kidx == 8) { kidx = 0; cin += 8; }
        else           { kidx += 1; cin += 7; }
    };
    auto storeB = [&](int buf, v8h_u v, int krow) {
        #pragma unroll
        for (int j = 0; j < 8; ++j)
            lB[buf][(pg + j) * TKS + krow] = v[j];
    };

    const int nstage = KTOT / TKS;        // 72

    // prologue: fill buffer 0 (stage 0)
    fillA(0, 0);
    {
        v8h_u v0 = loadRow(cinA, kidxA);
        v8h_u v1 = loadRow(cinB, kidxB);
        storeB(0, v0, br);
        storeB(0, v1, br + 32);
        advance(cinA, kidxA);
        advance(cinB, kidxB);
    }
    wait_async0();
    __syncthreads();

    const int am = mi * 32 + (lane & 15);
    const int ko = (lane >> 4) * 8;
    const int bn = ni * 32 + (lane & 15);
    const int bk = (lane >> 4) * 8;

    for (int s = 0; s < nstage; ++s) {
        const int cur = s & 1, nxt = cur ^ 1;
        const bool more = (s + 1 < nstage);

        v8h_u bp0 = {}, bp1 = {};
        if (more) {
            fillA(nxt, (s + 1) * TKS);    // async -> other buffer
            bp0 = loadRow(cinA, kidxA);   // global -> regs
            bp1 = loadRow(cinB, kidxB);
            advance(cinA, kidxA);
            advance(cinB, kidxB);
        }

        // two 32-K WMMA chunks from the current stage
        #pragma unroll
        for (int c = 0; c < 2; ++c) {
            const int c32 = c * 32;
            v16h a0 = mkfrag(*(const v8h*)&lA[cur][am * TKS + c32 + ko],
                             *(const v8h*)&lA[cur][am * TKS + c32 + 16 + ko]);
            v16h a1 = mkfrag(*(const v8h*)&lA[cur][(am + 16) * TKS + c32 + ko],
                             *(const v8h*)&lA[cur][(am + 16) * TKS + c32 + 16 + ko]);
            v16h b0 = mkfrag(*(const v8h*)&lB[cur][bn * TKS + c32 + bk],
                             *(const v8h*)&lB[cur][bn * TKS + c32 + 16 + bk]);
            v16h b1 = mkfrag(*(const v8h*)&lB[cur][(bn + 16) * TKS + c32 + bk],
                             *(const v8h*)&lB[cur][(bn + 16) * TKS + c32 + 16 + bk]);

            acc00 = __builtin_amdgcn_wmma_f32_16x16x32_f16(
                false, a0, false, b0, (short)0, acc00, false, false);
            acc01 = __builtin_amdgcn_wmma_f32_16x16x32_f16(
                false, a0, false, b1, (short)0, acc01, false, false);
            acc10 = __builtin_amdgcn_wmma_f32_16x16x32_f16(
                false, a1, false, b0, (short)0, acc10, false, false);
            acc11 = __builtin_amdgcn_wmma_f32_16x16x32_f16(
                false, a1, false, b1, (short)0, acc11, false, false);
        }

        if (more) {
            storeB(nxt, bp0, br);         // regs -> other buffer
            storeB(nxt, bp1, br + 32);
        }

        wait_async0();
        __syncthreads();
    }

    // store: VGPR i -> M = i (lanes<16) / 8+i (lanes>=16), N = lane&15
    const int Mb = cout0 + mi * 32 + ((lane >> 4) * 8);
    const int N  = lane & 15;
    float* ob = out + (size_t)b * COUT_ * NPIX + p0 + ni * 32 + N;
    #pragma unroll
    for (int i = 0; i < 8; ++i) {
        ob[(size_t)(Mb + i) * NPIX + 0]       = acc00[i];
        ob[(size_t)(Mb + i) * NPIX + 16]      = acc01[i];
        ob[(size_t)(Mb + 16 + i) * NPIX + 0]  = acc10[i];
        ob[(size_t)(Mb + 16 + i) * NPIX + 16] = acc11[i];
    }
}

// ---------------------------------------------------------------------------
extern "C" void kernel_launch(void* const* d_in, const int* in_sizes, int n_in,
                              void* d_out, int out_size, void* d_ws, size_t ws_size,
                              hipStream_t stream)
{
    (void)in_sizes; (void)n_in; (void)out_size; (void)ws_size;

    const float* x      = (const float*)d_in[0];
    const float* w      = (const float*)d_in[1];
    const float* mod_w  = (const float*)d_in[2];
    const float* mod_b  = (const float*)d_in[3];
    const float* ln_g   = (const float*)d_in[4];
    const float* ln_b   = (const float*)d_in[5];
    const float* weight = (const float*)d_in[6];
    const float* scale  = (const float*)d_in[7];
    float* out = (float*)d_out;

    // workspace layout
    char* ws = (char*)d_ws;
    float*    style = (float*)ws;                        // 32 KB
    _Float16* xp    = (_Float16*)(ws + 32768);           // 16*512*1156*2 = 18.9 MB
    _Float16* wmodh = (_Float16*)(ws + 32768 + (size_t)B_ * CIN_ * PIMG * 2);

    k_modulate<<<B_, 512, 0, stream>>>(w, mod_w, mod_b, ln_g, ln_b, scale, style);

    {
        int total = B_ * CIN_ * PIMG;
        k_pad<<<(total + 255) / 256, 256, 0, stream>>>(x, xp, total);
    }

    k_demod<<<B_ * COUT_, 256, 0, stream>>>(weight, style, wmodh);

    dim3 grid(NPIX / 64, COUT_ / 128, B_);
    k_conv_wmma<<<grid, 256, 0, stream>>>(wmodh, xp, out);
}
